// HybridQLSTM_65481071407478
// MI455X (gfx1250) — compile-verified, matrix operand order checked by reading, not run on hardware
//
#include <hip/hip_runtime.h>
#include <hip/hip_bf16.h>
#include <math.h>

// ---------------------------------------------------------------------------
// HybridQLSTM fused kernel for MI455X (gfx1250, wave32, WMMA).
//
// B=64, S=2048, K=2 (D=4), H=256, 4H=1024, NCLS=2.
//
// Phase A (pack): w_hh [1024,256], w1 [256,256], w2 [256,256]  -> bf16 WMMA
//                 B-fragment layout in d_ws (768 KB total).
// Phase B (fused): 4 workgroups x 512 threads (16 waves). Each workgroup owns
//                 16 batch rows; per timestep:
//                   1. gates_hh = h @ w_hh.T  via v_wmma_f32_16x16x32_bf16
//                      (16 waves x 4 ntiles x 8 ktiles = 512 WMMA),
//                      B streamed from L2 via global_load_b128 (pointers
//                      laundered as integers + explicit addrspace(1) so LICM
//                      can't hoist 256 VGPRs of weights into scratch, and the
//                      loads don't fall back to FLAT and tie up DScnt)
//                   2. elementwise LSTM cell (feat + gates_x on the fly, D=4),
//                      writes h back to LDS in bf16 A-frag layout
//                   3. h1 = relu(h@w1.T+b1)   (16 waves x 8 WMMA)
//                   4. h2 = relu(h1@w2.T+b2)  (16 waves x 8 WMMA, f32 -> LDS)
//                   5. logits = h2@w3.T+b3    (256-thread partial + 32 reduce)
// ---------------------------------------------------------------------------

typedef __attribute__((ext_vector_type(16))) __bf16 v16bf;
typedef __attribute__((ext_vector_type(8)))  float  v8f;

#define GLOBAL __attribute__((address_space(1)))

#define S_LEN   2048
#define HDIM    256
#define GDIM    1024   // 4*H
#define TPB     512

__device__ __forceinline__ unsigned short f2bf(float f) {
    unsigned u = __float_as_uint(f);
    unsigned r = u + 0x7FFFu + ((u >> 16) & 1u);   // round-to-nearest-even
    return (unsigned short)(r >> 16);
}
__device__ __forceinline__ float sigmoidf_(float v) {
    return 1.0f / (1.0f + __expf(-v));
}

// A/B 16-bit fragment addressing (16x32 A-tile / 32x16 B-tile), per ISA 7.12.2:
// lane = G*16 + m ; VGPR j: K = (j/4)*16 + G*8 + 2*(j%4) + p.
// Returns halfword offset within [ktile][lane][16 halfs] LDS image.
__device__ __forceinline__ int afrag_off(int m, int k) {
    int kt   = k >> 5;
    int kp   = k & 31;
    int half = kp >> 4;
    int r    = kp & 15;
    int G    = r >> 3;
    int jj   = (r & 7) >> 1;
    int p    = kp & 1;
    int lane = G * 16 + m;
    int dw   = half * 4 + jj;
    return kt * 512 + lane * 16 + dw * 2 + p;
}

// ---------------------------------------------------------------------------
// Pack a row-major weight W[Ntot, Ktot] (used as B[k,n] = W[n,k]) into bf16
// WMMA B-fragments: tile t = nt*ktiles + kt, 512 halfs per tile, laid out so
// lane L reads halfs [L*16 .. L*16+15] as its v16bf operand.
// ---------------------------------------------------------------------------
__global__ void pack_b_frags(const float* __restrict__ W,
                             unsigned short* __restrict__ dst,
                             int ntiles, int ktiles, int Ktot) {
    int idx   = blockIdx.x * blockDim.x + threadIdx.x;
    int total = ntiles * ktiles * 512;
    if (idx >= total) return;
    int t    = idx >> 9;
    int e    = idx & 511;
    int nt   = t / ktiles;
    int kt   = t - nt * ktiles;
    int lane = e >> 4;
    int hw   = e & 15;
    int dw   = hw >> 1;
    int p    = hw & 1;
    int G    = lane >> 4;
    int n_   = lane & 15;
    int half = dw >> 2;
    int jj   = dw & 3;
    int kp   = half * 16 + G * 8 + jj * 2 + p;
    int n    = nt * 16 + n_;
    int k    = kt * 32 + kp;
    dst[idx] = f2bf(W[n * Ktot + k]);
}

// ---------------------------------------------------------------------------
// Fused LSTM + classifier. grid = 4 (batch tiles of 16 rows), block = 512.
// ---------------------------------------------------------------------------
__global__ __launch_bounds__(TPB, 1)
void qlstm_fused(const float* __restrict__ x,        // [64, 2048, 4]
                 const float* __restrict__ conv_w,   // [4]
                 const float* __restrict__ conv_b,   // [4]
                 const float* __restrict__ w_ih,     // [1024, 4]
                 const float* __restrict__ b_ih,     // [1024]
                 const float* __restrict__ b_hh,     // [1024]
                 const float* __restrict__ b1,       // [256]
                 const float* __restrict__ b2,       // [256]
                 const float* __restrict__ w3,       // [2, 256]
                 const float* __restrict__ b3,       // [2]
                 const unsigned short* __restrict__ whh_pk, // 64x8 tiles
                 const unsigned short* __restrict__ w1_pk,  // 16x8 tiles
                 const unsigned short* __restrict__ w2_pk,  // 16x8 tiles
                 float* __restrict__ out)            // [64, 2048, 2]
{
    __shared__ __align__(32) unsigned short s_afrag[8 * 32 * 16];  // h  (bf16 A-frags)
    __shared__ __align__(32) unsigned short s_h1frag[8 * 32 * 16]; // h1 (bf16 A-frags)
    __shared__ float s_gates[16 * GDIM];   // raw gates f32 (64 KB); first
                                           // 16*256 reused as h2 (f32) later
    __shared__ float s_c[16 * HDIM];       // cell state, f32 (16 KB)
    __shared__ float s_wih[GDIM * 4];      // w_ih (16 KB)
    __shared__ float s_bias[GDIM];         // b_ih + b_hh
    __shared__ float s_b1[HDIM];
    __shared__ float s_b2[HDIM];
    __shared__ float s_w3[2 * HDIM];
    __shared__ float s_b3[2];
    __shared__ float s_red[16 * 2 * 8];    // head partial sums

    const int tile = blockIdx.x;          // batch rows [tile*16, tile*16+16)
    const int tid  = threadIdx.x;
    const int wave = tid >> 5;
    const int lane = tid & 31;

    // ---- one-time LDS init -------------------------------------------------
    for (int i = tid; i < GDIM * 4; i += TPB) s_wih[i] = w_ih[i];
    for (int i = tid; i < GDIM; i += TPB)     s_bias[i] = b_ih[i] + b_hh[i];
    for (int i = tid; i < HDIM; i += TPB)   { s_b1[i] = b1[i]; s_b2[i] = b2[i]; }
    for (int i = tid; i < 2 * HDIM; i += TPB) s_w3[i] = w3[i];
    if (tid < 2) s_b3[tid] = b3[tid];
    for (int i = tid; i < 8 * 32 * 16; i += TPB) s_afrag[i] = 0;   // h = 0
    for (int i = tid; i < 16 * HDIM; i += TPB)   s_c[i] = 0.0f;    // c = 0

    float cw[4], cb[4];
#pragma unroll
    for (int d = 0; d < 4; ++d) { cw[d] = conv_w[d]; cb[d] = conv_b[d]; }

    __syncthreads();

    const int lm = lane >> 4;     // C-layout row-group
    const int ln = lane & 15;     // C-layout column within tile

    for (int s = 0; s < S_LEN; ++s) {
        // Launder the packed-weight pointers as integers (loop-carried dep so
        // LICM cannot hoist 32 B-fragments / 256 VGPRs out of the loop and
        // spill them), then rebuild them as explicit global (addrspace(1))
        // pointers so the loads lower to global_load_b128 (LOADcnt only),
        // not flat_load (which would also tick DScnt and contend with the
        // ds_load_b128 A-fragment traffic).
        unsigned long long whh_a = (unsigned long long)whh_pk;
        unsigned long long w1_a  = (unsigned long long)w1_pk;
        unsigned long long w2_a  = (unsigned long long)w2_pk;
        asm volatile("" : "+s"(whh_a), "+s"(w1_a), "+s"(w2_a));
        GLOBAL const unsigned short* whh_p = (GLOBAL const unsigned short*)whh_a;
        GLOBAL const unsigned short* w1_p  = (GLOBAL const unsigned short*)w1_a;
        GLOBAL const unsigned short* w2_p  = (GLOBAL const unsigned short*)w2_a;

        // ---- phase 1: gates_hh = h @ w_hh.T (bf16 WMMA, f32 accum) -------
        // wave handles N-tiles [wave*4, wave*4+4)
        {
            v8f acc[4] = {};
#pragma unroll
            for (int kt = 0; kt < 8; ++kt) {
                v16bf a = *(const v16bf*)(s_afrag + kt * 512 + lane * 16);
#pragma unroll
                for (int t = 0; t < 4; ++t) {
                    int nt = wave * 4 + t;
                    v16bf b = *(GLOBAL const v16bf*)(whh_p + ((nt * 8 + kt) * 512 + lane * 16));
                    acc[t] = __builtin_amdgcn_wmma_f32_16x16x32_bf16(
                        false, a, false, b, (short)0, acc[t], false, false);
                }
            }
#pragma unroll
            for (int t = 0; t < 4; ++t) {
                int n0 = (wave * 4 + t) * 16 + ln;
#pragma unroll
                for (int r = 0; r < 8; ++r)
                    s_gates[(r + 8 * lm) * GDIM + n0] = acc[t][r];
            }
        }
        __syncthreads();

        // ---- phase 2: elementwise LSTM cell (feat + gates_x on the fly) --
        {
            int m  = tid >> 5;            // wave w owns batch row m = w
            int j0 = (tid & 31) * 8;      // 8 consecutive hidden units
            const float* xrow = x + (((size_t)(tile * 16 + m) * S_LEN + s) << 2);
            const float4 xv = *(const float4*)xrow;            // broadcast row
            if (lane == 0 && s + 1 < S_LEN) __builtin_prefetch(xrow + 4, 0, 3);
            float f0 = sigmoidf_(xv.x * cw[0] + cb[0]);
            float f1 = sigmoidf_(xv.y * cw[1] + cb[1]);
            float f2 = sigmoidf_(xv.z * cw[2] + cb[2]);
            float f3 = sigmoidf_(xv.w * cw[3] + cb[3]);
#pragma unroll
            for (int q = 0; q < 8; ++q) {
                int j = j0 + q;
                float gi = s_bias[j]       + f0 * s_wih[ j        * 4 + 0] + f1 * s_wih[ j        * 4 + 1] + f2 * s_wih[ j        * 4 + 2] + f3 * s_wih[ j        * 4 + 3];
                float gf = s_bias[256 + j] + f0 * s_wih[(256 + j) * 4 + 0] + f1 * s_wih[(256 + j) * 4 + 1] + f2 * s_wih[(256 + j) * 4 + 2] + f3 * s_wih[(256 + j) * 4 + 3];
                float gg = s_bias[512 + j] + f0 * s_wih[(512 + j) * 4 + 0] + f1 * s_wih[(512 + j) * 4 + 1] + f2 * s_wih[(512 + j) * 4 + 2] + f3 * s_wih[(512 + j) * 4 + 3];
                float go = s_bias[768 + j] + f0 * s_wih[(768 + j) * 4 + 0] + f1 * s_wih[(768 + j) * 4 + 1] + f2 * s_wih[(768 + j) * 4 + 2] + f3 * s_wih[(768 + j) * 4 + 3];
                float ig = sigmoidf_(s_gates[m * GDIM + j]       + gi);
                float fg = sigmoidf_(s_gates[m * GDIM + 256 + j] + gf);
                float gc = tanhf    (s_gates[m * GDIM + 512 + j] + gg);
                float og = sigmoidf_(s_gates[m * GDIM + 768 + j] + go);
                float c  = fg * s_c[m * HDIM + j] + ig * gc;
                s_c[m * HDIM + j] = c;
                float h = og * tanhf(c);
                s_afrag[afrag_off(m, j)] = f2bf(h);
            }
        }
        __syncthreads();

        // ---- phase 3: h1 = relu(h @ w1.T + b1) (wave -> ntile wave) ------
        {
            v8f acc = {};
#pragma unroll
            for (int kt = 0; kt < 8; ++kt) {
                v16bf a = *(const v16bf*)(s_afrag + kt * 512 + lane * 16);
                v16bf b = *(GLOBAL const v16bf*)(w1_p + ((wave * 8 + kt) * 512 + lane * 16));
                acc = __builtin_amdgcn_wmma_f32_16x16x32_bf16(
                    false, a, false, b, (short)0, acc, false, false);
            }
            int n = wave * 16 + ln;
#pragma unroll
            for (int r = 0; r < 8; ++r) {
                int   m = r + 8 * lm;
                float v = acc[r] + s_b1[n];
                s_h1frag[afrag_off(m, n)] = f2bf(v > 0.0f ? v : 0.0f);
            }
        }
        __syncthreads();

        // ---- phase 4: h2 = relu(h1 @ w2.T + b2), stored f32 in LDS -------
        {
            float* s_h2val = s_gates;      // reuse (gates dead after phase 2)
            v8f acc = {};
#pragma unroll
            for (int kt = 0; kt < 8; ++kt) {
                v16bf a = *(const v16bf*)(s_h1frag + kt * 512 + lane * 16);
                v16bf b = *(GLOBAL const v16bf*)(w2_p + ((wave * 8 + kt) * 512 + lane * 16));
                acc = __builtin_amdgcn_wmma_f32_16x16x32_bf16(
                    false, a, false, b, (short)0, acc, false, false);
            }
            int n = wave * 16 + ln;
#pragma unroll
            for (int r = 0; r < 8; ++r) {
                int   m = r + 8 * lm;
                float v = acc[r] + s_b2[n];
                s_h2val[m * HDIM + n] = v > 0.0f ? v : 0.0f;
            }
        }
        __syncthreads();

        // ---- phase 5: logits = h2 @ w3.T + b3 ----------------------------
        // 5a: 256 threads compute (m, cls, chunk-of-32) partials
        {
            const float* s_h2val = s_gates;
            if (tid < 256) {
                int m = tid >> 4, cls = (tid >> 3) & 1, ch = tid & 7;
                float sum = 0.0f;
#pragma unroll 8
                for (int kk = 0; kk < 32; ++kk) {
                    int k = ch * 32 + kk;
                    sum += s_h2val[m * HDIM + k] * s_w3[cls * HDIM + k];
                }
                s_red[(m * 2 + cls) * 8 + ch] = sum;
            }
        }
        __syncthreads();
        // 5b: 32 threads reduce 8 partials each and store the logits
        if (tid < 32) {
            int m = tid >> 1, cls = tid & 1;
            float sum = s_b3[cls];
#pragma unroll
            for (int ch = 0; ch < 8; ++ch) sum += s_red[(m * 2 + cls) * 8 + ch];
            out[(((size_t)(tile * 16 + m)) * S_LEN + s) * 2 + cls] = sum;
        }
        __syncthreads();   // protects s_gates/s_red before next phase 1
    }
}

// ---------------------------------------------------------------------------
extern "C" void kernel_launch(void* const* d_in, const int* in_sizes, int n_in,
                              void* d_out, int out_size, void* d_ws, size_t ws_size,
                              hipStream_t stream) {
    const float* x      = (const float*)d_in[0];
    const float* conv_w = (const float*)d_in[1];
    const float* conv_b = (const float*)d_in[2];
    const float* w_ih   = (const float*)d_in[3];
    const float* w_hh   = (const float*)d_in[4];
    const float* b_ih   = (const float*)d_in[5];
    const float* b_hh   = (const float*)d_in[6];
    const float* w1     = (const float*)d_in[7];
    const float* b1     = (const float*)d_in[8];
    const float* w2     = (const float*)d_in[9];
    const float* b2     = (const float*)d_in[10];
    const float* w3     = (const float*)d_in[11];
    const float* b3     = (const float*)d_in[12];
    float* out = (float*)d_out;

    // workspace layout (bf16 halves): w_hh frags | w1 frags | w2 frags
    unsigned short* whh_pk = (unsigned short*)d_ws;        // 64*8*512 = 262144
    unsigned short* w1_pk  = whh_pk + 64 * 8 * 512;        // 16*8*512 =  65536
    unsigned short* w2_pk  = w1_pk  + 16 * 8 * 512;        // 16*8*512 =  65536

    pack_b_frags<<<(64 * 8 * 512 + 255) / 256, 256, 0, stream>>>(w_hh, whh_pk, 64, 8, HDIM);
    pack_b_frags<<<(16 * 8 * 512 + 255) / 256, 256, 0, stream>>>(w1,  w1_pk, 16, 8, HDIM);
    pack_b_frags<<<(16 * 8 * 512 + 255) / 256, 256, 0, stream>>>(w2,  w2_pk, 16, 8, HDIM);

    qlstm_fused<<<4, TPB, 0, stream>>>(x, conv_w, conv_b, w_ih, b_ih, b_hh,
                                       b1, b2, w3, b3, whh_pk, w1_pk, w2_pk, out);
}
